// FeatureExtractor_79250736546302
// MI455X (gfx1250) — compile-verified, matrix operand order checked by reading, not run on hardware
//
#include <hip/hip_runtime.h>
#include <hip/hip_bf16.h>
#include <climits>

typedef __bf16 bf16_t;
typedef bf16_t v16bf __attribute__((ext_vector_type(16)));
typedef bf16_t v8bf  __attribute__((ext_vector_type(8)));
typedef float  v8f   __attribute__((ext_vector_type(8)));

#define T_LEN     2048          // samples per row (K dim)
#define NCOLS     2048          // 1024 bins x (re,im) interleaved (N dim)
#define M_TILE    64            // rows per block
#define A_STRIDE  2056          // 2048 + 8 bf16 pad -> LDS bank spread
#define GEMM_LDS  (M_TILE*A_STRIDE*2 + 8*M_TILE*4)   // A slab + per-wave row partials
#define FEAT_LDS  (8*T_LEN*4)   // 8 cached rows
#define FS_F      30.0f

__device__ __forceinline__ v8f vzero8() {
  v8f z;
#pragma unroll
  for (int i = 0; i < 8; ++i) z[i] = 0.f;
  return z;
}

// Build a 16x32 bf16 WMMA fragment for this lane from two contiguous 8-elem runs
// (K = base+[0..7] and base+16+[0..7]) per the CDNA5 16-bit A/B VGPR layout.
__device__ __forceinline__ v16bf frag16(const bf16_t* p) {
  v8bf lo = *(const v8bf*)p;
  v8bf hi = *(const v8bf*)(p + 16);
  return __builtin_shufflevector(lo, hi, 0,1,2,3,4,5,6,7,8,9,10,11,12,13,14,15);
}

// ---------------- twiddle matrix: Wt[n][t], n even -> cos, n odd -> -sin ----
__global__ void dft_twiddle_kernel(bf16_t* __restrict__ Wt) {
  int idx = blockIdx.x * blockDim.x + threadIdx.x;   // 0 .. 2048*2048-1
  int n = idx >> 11;
  int t = idx & (T_LEN - 1);
  int bin = n >> 1;
  int ph = (bin * t) & (T_LEN - 1);                  // exact phase reduction
  float ang = (float)ph * (6.28318530717958647692f / (float)T_LEN);
  float v = (n & 1) ? -sinf(ang) : cosf(ang);
  Wt[idx] = (bf16_t)v;
}

// -------- fused DFT-magnitude GEMM: rowS[r] = sum_{k=0..1023} |X_k(r)| ------
__global__ void dft_mag_gemm_kernel(const float* __restrict__ x,
                                    const bf16_t* __restrict__ Wt,
                                    float* __restrict__ rowS) {
  extern __shared__ char smem[];
  bf16_t* As   = (bf16_t*)smem;                       // 64 x 2056 bf16
  float*  part = (float*)(smem + M_TILE*A_STRIDE*2);  // [8 waves][64 rows]
  const int tid  = threadIdx.x;                       // 256 threads = 8 waves
  const int lane = tid & 31;
  const int wv   = tid >> 5;
  const long rowBase = (long)blockIdx.x * M_TILE;

  // Stage A slab: f32 -> bf16 into LDS (x read from HBM exactly once)
  for (int i = tid; i < M_TILE * (T_LEN / 4); i += 256) {
    int r  = i >> 9;
    int c4 = i & 511;
    float4 v = ((const float4*)(x + (rowBase + r) * T_LEN))[c4];
    bf16_t* d = As + r * A_STRIDE + c4 * 4;
    d[0] = (bf16_t)v.x; d[1] = (bf16_t)v.y; d[2] = (bf16_t)v.z; d[3] = (bf16_t)v.w;
  }
  __syncthreads();

  const int hs  = lane >> 4;      // lane half selects K group (+8 / +24)
  const int l15 = lane & 15;
  float rowpart[4][8] = {};       // per-lane row partial |X| sums

  for (int slice = 0; slice < 8; ++slice) {
    const int colBase = slice * 256 + wv * 32;   // this wave's 32 spectrum cols
    v8f acc[4][2];
#pragma unroll
    for (int rt = 0; rt < 4; ++rt)
#pragma unroll
      for (int ct = 0; ct < 2; ++ct) acc[rt][ct] = vzero8();

    for (int k = 0; k < T_LEN; k += 32) {
      const int klo = k + hs * 8;
      v16bf a[4], b[2];
#pragma unroll
      for (int rt = 0; rt < 4; ++rt)          // A from LDS (ds_load_b128 x2)
        a[rt] = frag16(As + (16 * rt + l15) * A_STRIDE + klo);
#pragma unroll
      for (int ct = 0; ct < 2; ++ct)          // B from L2-resident Wt (transposed)
        b[ct] = frag16(Wt + (long)(colBase + 16 * ct + l15) * T_LEN + klo);
#pragma unroll
      for (int rt = 0; rt < 4; ++rt)
#pragma unroll
        for (int ct = 0; ct < 2; ++ct)
          acc[rt][ct] = __builtin_amdgcn_wmma_f32_16x16x32_bf16(
              false, a[rt], false, b[ct], (short)0, acc[rt][ct], false, false);
    }

    // Epilogue: pair (re,im) across adjacent lanes, accumulate |X| per row.
#pragma unroll
    for (int rt = 0; rt < 4; ++rt)
#pragma unroll
      for (int j = 0; j < 8; ++j) {
#pragma unroll
        for (int ct = 0; ct < 2; ++ct) {
          float v = acc[rt][ct][j];           // D: lane N=l15, row M=j+8*hs
          float p = v * v;
          p += __shfl_xor(p, 1, 32);          // re^2 + im^2 (cols 2b, 2b+1)
          rowpart[rt][j] += sqrtf(p);         // duplicated in both pair lanes
        }
      }
  }

  // Reduce over the 16 N-lanes of each half (each bin counted twice -> *0.5),
  // write deterministic per-wave partials, then ordered cross-wave sum.
#pragma unroll
  for (int rt = 0; rt < 4; ++rt)
#pragma unroll
    for (int j = 0; j < 8; ++j) {
      float r = rowpart[rt][j];
      r += __shfl_xor(r, 1, 32);
      r += __shfl_xor(r, 2, 32);
      r += __shfl_xor(r, 4, 32);
      r += __shfl_xor(r, 8, 32);
      r *= 0.5f;
      if (l15 == 0) part[wv * M_TILE + 16 * rt + 8 * hs + j] = r;
    }
  __syncthreads();
  if (tid < M_TILE) {
    float s = 0.f;
#pragma unroll
    for (int w = 0; w < 8; ++w) s += part[w * M_TILE + tid];
    rowS[rowBase + tid] = s;
  }
}

// ---------------- per-row scan features (one wave per row) ------------------
__global__ void feature_kernel(const float* __restrict__ x,
                               const float* __restrict__ rowS,
                               float* __restrict__ out) {
  extern __shared__ char smem[];
  float* srows = (float*)smem;
  const int lane = threadIdx.x & 31;
  const int wv   = threadIdx.x >> 5;
  const long row = (long)blockIdx.x * 8 + wv;
  const float* xr = x + row * T_LEN;
  float* s = srows + wv * T_LEN;
  for (int i4 = lane; i4 < T_LEN / 4; i4 += 32)
    ((float4*)s)[i4] = ((const float4*)xr)[i4];
  __syncthreads();

  float mx = -3.4e38f, mn = 3.4e38f, sum = 0.f, ssq = 0.f, alt = 0.f;
  int cp = 0, cv = 0;
  int pmin = INT_MAX, va = INT_MAX, vb = INT_MAX, vmax = -1;
  for (int i = lane; i < T_LEN; i += 32) {
    float m = s[i];
    mx = fmaxf(mx, m); mn = fminf(mn, m);
    sum += m; ssq += m * m;
    alt += (i & 1) ? -m : m;                 // X_{N/2} = sum (-1)^t x_t
    if (i >= 1 && i <= T_LEN - 2) {
      float l = s[i - 1], r = s[i + 1];
      if (l < m && m > r) { cp++; pmin = min(pmin, i); }
      if (l > m && m < r) {
        cv++;
        if (i < va) { vb = va; va = i; }
        else if (i < vb) vb = i;
        vmax = max(vmax, i);
      }
    }
  }
#pragma unroll
  for (int off = 16; off > 0; off >>= 1) {
    mx = fmaxf(mx, __shfl_xor(mx, off, 32));
    mn = fminf(mn, __shfl_xor(mn, off, 32));
    sum += __shfl_xor(sum, off, 32);
    ssq += __shfl_xor(ssq, off, 32);
    alt += __shfl_xor(alt, off, 32);
    cp  += __shfl_xor(cp,  off, 32);
    cv  += __shfl_xor(cv,  off, 32);
    pmin = min(pmin, __shfl_xor(pmin, off, 32));
    int oa = __shfl_xor(va, off, 32);
    int ob = __shfl_xor(vb, off, 32);
    int na = min(va, oa);
    int nb = min(min(vb, ob), max(va, oa));   // two-smallest merge
    va = na; vb = nb;
    vmax = max(vmax, __shfl_xor(vmax, off, 32));
  }
  const int p0 = (pmin == INT_MAX) ? 1 : pmin;     // argmax-of-empty -> 0, +1
  const int v0 = (va == INT_MAX) ? 1 : va;
  const int v1 = (vb == INT_MAX) ? 1 : vb;
  const int vlast = (vmax < 0) ? (T_LEN - 2) : vmax;
  const float hlf = 0.5f * (s[p0] + s[v0]);

  float pa = 0.f, a2 = 0.f, a1 = 0.f;
  int li = INT_MAX, ri = -1;
  for (int i = lane; i < T_LEN; i += 32) {
    float si = s[i];
    if (i < T_LEN - 1) {
      float pr = 0.5f * (si + s[i + 1]);
      if (i >= v0 && i <= vlast - 2) pa += pr;
      if (i >= p0 && i <= v1 - 2)    a2 += pr;
      if (i >= v0 && i <= p0 - 2)    a1 += pr;
    }
    if (si >= hlf) {
      if (i >= v0 && i <  p0) li = min(li, i);
      if (i >  v0 && i <= p0) ri = max(ri, i);
    }
  }
#pragma unroll
  for (int off = 16; off > 0; off >>= 1) {
    pa += __shfl_xor(pa, off, 32);
    a2 += __shfl_xor(a2, off, 32);
    a1 += __shfl_xor(a1, off, 32);
    li = min(li, __shfl_xor(li, off, 32));
    ri = max(ri, __shfl_xor(ri, off, 32));
  }
  if (lane == 0) {
    const float n = (float)T_LEN;
    float var = (ssq - sum * sum / n) / (n - 1.f);
    float sd = sqrtf(var);
    // mean|FFT| = (2*sum_{0..1023}|X_k| - |X_0| + |X_1024|) / 2048
    float mean_amp = (2.f * rowS[row] - fabsf(sum) + fabsf(alt)) / n;
    bool gate = (cp >= 1) && (cv >= 2);
    int left  = (li == INT_MAX) ? v0 : li;
    int right = (ri < 0) ? p0 : ri;
    float PA = gate ? pa / FS_F : 0.f;
    float A2 = gate ? a2 / FS_F : 0.f;
    float PH = gate ? (s[p0] - s[v0]) : 0.f;
    float A1 = gate ? a1 / FS_F : 0.f;
    float PW = gate ? (float)(right - left) / FS_F : 0.f;
    float* o = out + row * 10;
    o[0] = mx; o[1] = mx - mn; o[2] = var; o[3] = sd; o[4] = mean_amp;
    o[5] = PA; o[6] = A2; o[7] = PH; o[8] = A1; o[9] = PW;
  }
}

extern "C" void kernel_launch(void* const* d_in, const int* in_sizes, int n_in,
                              void* d_out, int out_size, void* d_ws, size_t ws_size,
                              hipStream_t stream) {
  (void)in_sizes; (void)n_in; (void)out_size; (void)ws_size;
  const float* x = (const float*)d_in[0];
  float* out = (float*)d_out;
  bf16_t* Wt = (bf16_t*)d_ws;                                   // 8 MB twiddles
  float* rowS = (float*)((char*)d_ws + (size_t)NCOLS * T_LEN * sizeof(bf16_t));

  hipFuncSetAttribute(reinterpret_cast<const void*>(dft_mag_gemm_kernel),
                      hipFuncAttributeMaxDynamicSharedMemorySize, GEMM_LDS);
  hipFuncSetAttribute(reinterpret_cast<const void*>(feature_kernel),
                      hipFuncAttributeMaxDynamicSharedMemorySize, FEAT_LDS);

  dft_twiddle_kernel<<<(NCOLS * T_LEN) / 256, 256, 0, stream>>>(Wt);
  dft_mag_gemm_kernel<<<32768 / M_TILE, 256, GEMM_LDS, stream>>>(x, Wt, rowS);
  feature_kernel<<<32768 / 8, 256, FEAT_LDS, stream>>>(x, rowS, out);
}
